// Env_10883447128375
// MI455X (gfx1250) — compile-verified, matrix operand order checked by reading, not run on hardware
//
#include <hip/hip_runtime.h>
#include <hip/hip_bf16.h>
#include <stdint.h>

#define E_DIM 2048
#define H_DIM 2048
#define G_DIM 8192      // 4*H
#define L_SEQ 100
#define TEMP_D 256
#define NLAB 3
#define NB 65           // 64 WMMA blocks + 1 scalar (policy MLP) block
#define BS 256

typedef __attribute__((ext_vector_type(16))) __bf16 v16bf;
typedef __attribute__((ext_vector_type(8)))  float  v8f;

__device__ __forceinline__ unsigned short f2bf_bits(float f) {
  union { float f; unsigned u; } a; a.f = f;
  unsigned u = a.u;
  return (unsigned short)((u + 0x7FFFu + ((u >> 16) & 1u)) >> 16);  // RNE, finite inputs
}

__device__ __forceinline__ float sigf(float x) { return 1.0f / (1.0f + expf(-x)); }

// ---------------- device-wide split barrier (monotonic counter) ----------------
__device__ __forceinline__ void gsync(unsigned* cnt, unsigned& gen) {
  __threadfence();
  __syncthreads();
  if (threadIdx.x == 0) {
    __hip_atomic_fetch_add(cnt, 1u, __ATOMIC_ACQ_REL, __HIP_MEMORY_SCOPE_AGENT);
    gen += NB;
    while (__hip_atomic_load(cnt, __ATOMIC_ACQUIRE, __HIP_MEMORY_SCOPE_AGENT) < gen)
      __builtin_amdgcn_s_sleep(1);
  }
  __syncthreads();
  __threadfence();
}

// ---------------- WMMA mat-vec: gates = W_ih@x + W_hh@h + b_ih + b_hh ----------
// One wave owns 16 consecutive gate rows over K=2048 for both matrices.
// A = bf16 weights (16x32 tile), B = bf16 vector replicated into all 16 columns
// (same per-lane K mapping as A), D column 0 read from lanes 0 / 16.
// Inner loop is pure b128 loads + v_wmma: all conversions are precomputed.
__device__ void gates_wmma(const unsigned short* __restrict__ Wih,
                           const unsigned short* __restrict__ Whh,
                           const unsigned short* __restrict__ xbf,
                           const unsigned short* __restrict__ hbf,
                           const float* __restrict__ b_ih,
                           const float* __restrict__ b_hh,
                           float* __restrict__ gates) {
  const int lane    = threadIdx.x & 31;
  const int wave    = (int)((blockIdx.x * blockDim.x + threadIdx.x) >> 5); // 0..511
  const int rowbase = wave * 16;
  const int r       = rowbase + (lane & 15);
  const int khalf   = (lane >> 4) ? 8 : 0;   // lanes>=16 hold K 8..15 / 24..31

  v8f acc = {};

  const unsigned short* WrowI = Wih + (size_t)r * E_DIM + khalf;
  const unsigned short* WrowH = Whh + (size_t)r * H_DIM + khalf;
  const unsigned short* xp    = xbf + khalf;
  const unsigned short* hp    = hbf + khalf;

#pragma unroll 2
  for (int kb = 0; kb < E_DIM; kb += 32) {
    union { v16bf v; uint4 u[2]; } a, b;
    a.u[0] = *(const uint4*)(WrowI + kb);
    a.u[1] = *(const uint4*)(WrowI + kb + 16);
    b.u[0] = *(const uint4*)(xp + kb);
    b.u[1] = *(const uint4*)(xp + kb + 16);
    __builtin_prefetch(WrowI + kb + 512, 0, 3);
    acc = __builtin_amdgcn_wmma_f32_16x16x32_bf16(false, a.v, false, b.v,
                                                  (short)0, acc, false, false);
  }
#pragma unroll 2
  for (int kb = 0; kb < H_DIM; kb += 32) {
    union { v16bf v; uint4 u[2]; } a, b;
    a.u[0] = *(const uint4*)(WrowH + kb);
    a.u[1] = *(const uint4*)(WrowH + kb + 16);
    b.u[0] = *(const uint4*)(hp + kb);
    b.u[1] = *(const uint4*)(hp + kb + 16);
    __builtin_prefetch(WrowH + kb + 512, 0, 3);
    acc = __builtin_amdgcn_wmma_f32_16x16x32_bf16(false, a.v, false, b.v,
                                                  (short)0, acc, false, false);
  }

  if (lane == 0) {
#pragma unroll
    for (int j = 0; j < 8; ++j) {
      int rr = rowbase + j;
      gates[rr] = acc[j] + b_ih[rr] + b_hh[rr];
    }
  } else if (lane == 16) {
#pragma unroll
    for (int j = 0; j < 8; ++j) {
      int rr = rowbase + 8 + j;
      gates[rr] = acc[j] + b_ih[rr] + b_hh[rr];
    }
  }
}

// ---------------- policy MLP on the scalar block -------------------------------
__device__ void policy_scalar(const float* __restrict__ c, const float* __restrict__ h,
                              const float* __restrict__ wvec, const float* __restrict__ tvec,
                              const float* __restrict__ W_s2t, const float* __restrict__ b_s2t,
                              const float* __restrict__ W_t2o, const float* __restrict__ b_t2o,
                              float* __restrict__ ps_t, float* __restrict__ acts_t,
                              unsigned* __restrict__ keep) {
  __shared__ float tmp[TEMP_D];
  const int r = threadIdx.x;                 // one temp row per thread
  const float* Wr = W_s2t + (size_t)r * (2 * H_DIM + 2 * E_DIM);
  const float* segs[4] = { c, h, wvec, tvec };   // state = [c, h, w, t]
  float s = 0.f;
#pragma unroll
  for (int sg = 0; sg < 4; ++sg) {
    const float* v = segs[sg];
    const float* W2 = Wr + sg * 2048;
    for (int j = 0; j < 2048; j += 4) {
      float4 a4 = *(const float4*)(W2 + j);
      float4 v4 = *(const float4*)(v + j);
      s = fmaf(a4.x, v4.x, s); s = fmaf(a4.y, v4.y, s);
      s = fmaf(a4.z, v4.z, s); s = fmaf(a4.w, v4.w, s);
    }
  }
  tmp[r] = tanhf(s + b_s2t[r]);
  __syncthreads();
  if (threadIdx.x == 0) {
    float l0 = b_t2o[0], l1 = b_t2o[1];
    for (int j = 0; j < TEMP_D; ++j) {
      l0 = fmaf(W_t2o[j], tmp[j], l0);
      l1 = fmaf(W_t2o[TEMP_D + j], tmp[j], l1);
    }
    float s0 = sigf(l0), s1 = sigf(l1);
    float mx = fmaxf(s0, s1);
    float e0 = expf(s0 - mx), e1 = expf(s1 - mx);
    float inv = 1.0f / (e0 + e1);
    float p0 = e0 * inv, p1 = e1 * inv;
    ps_t[0] = p0; ps_t[1] = p1;
    int a = (p1 > p0) ? 1 : 0;               // argmax, first index wins ties
    *acts_t = (float)a;
    *(volatile unsigned*)keep = (unsigned)a; // published by gsync fences
  }
}

// ---------------- persistent scan kernel ----------------------------------------
__global__ void k_persist(const unsigned short* wih, const unsigned short* whh,
                          float* h, float* c, unsigned short* hbf, float* gates,
                          const unsigned short* sents_bf, const unsigned short* words_bf,
                          const float* words, const float* tgts,
                          const float* b_ih, const float* b_hh,
                          const float* W_s2t, const float* b_s2t,
                          const float* W_t2o, const float* b_t2o,
                          float* ps, float* acts, unsigned* ctrl) {
  unsigned gen = 0;
  unsigned* cnt = &ctrl[0];
  unsigned* keep = &ctrl[1];
  const bool scalarBlk = (blockIdx.x == NB - 1);
  const int gid = (int)(blockIdx.x * blockDim.x + threadIdx.x);

  // ---- encoder scan ----
  for (int t = 0; t < L_SEQ; ++t) {
    if (!scalarBlk)
      gates_wmma(wih, whh, sents_bf + (size_t)t * E_DIM, hbf, b_ih, b_hh, gates);
    gsync(cnt, gen);
    if (gid < H_DIM) {
      float gi = gates[gid], gf = gates[H_DIM + gid];
      float gg = gates[2 * H_DIM + gid], go = gates[3 * H_DIM + gid];
      float c2 = sigf(gf) * c[gid] + sigf(gi) * tanhf(gg);
      float hn = sigf(go) * tanhf(c2);
      c[gid] = c2;
      h[gid] = hn;
      hbf[gid] = f2bf_bits(hn);
    }
    gsync(cnt, gen);
  }

  // ---- decoder scan: WMMA gates and policy MLP run concurrently ----
  for (int t = 0; t < L_SEQ; ++t) {
    if (!scalarBlk)
      gates_wmma(wih, whh, words_bf + (size_t)t * E_DIM, hbf, b_ih, b_hh, gates);
    else
      policy_scalar(c, h, words + (size_t)t * E_DIM, tgts + (size_t)t * E_DIM,
                    W_s2t, b_s2t, W_t2o, b_t2o, ps + 2 * t, acts + t, keep);
    gsync(cnt, gen);
    unsigned kf = *(volatile unsigned*)keep;
    if (gid < H_DIM && kf) {
      float gi = gates[gid], gf = gates[H_DIM + gid];
      float gg = gates[2 * H_DIM + gid], go = gates[3 * H_DIM + gid];
      float c2 = sigf(gf) * c[gid] + sigf(gi) * tanhf(gg);
      float hn = sigf(go) * tanhf(c2);
      c[gid] = c2;
      h[gid] = hn;
      hbf[gid] = f2bf_bits(hn);
    }
    gsync(cnt, gen);
  }
}

// ---------------- prep: fp32 -> bf16 weights ------------------------------------
__global__ void k_convert(const float* __restrict__ a, const float* __restrict__ b,
                          unsigned short* __restrict__ oa, unsigned short* __restrict__ ob) {
  size_t n = (size_t)G_DIM * E_DIM;
  size_t stride = (size_t)gridDim.x * blockDim.x;
  for (size_t i = blockIdx.x * (size_t)blockDim.x + threadIdx.x; i < n; i += stride) {
    oa[i] = f2bf_bits(a[i]);
    ob[i] = f2bf_bits(b[i]);
  }
}

// ---------------- prep: words / tgts / bf16 inputs / zero state -----------------
__global__ void k_prepvec(const float* __restrict__ x, const float* __restrict__ m,
                          const float* __restrict__ pe,
                          float* __restrict__ words, float* __restrict__ tgts,
                          unsigned short* __restrict__ sents_bf,
                          unsigned short* __restrict__ words_bf,
                          float* __restrict__ h, float* __restrict__ c,
                          unsigned short* __restrict__ hbf) {
  int d = (int)(blockIdx.x * blockDim.x + threadIdx.x);
  if (d >= E_DIM) return;
  float msum = 0.f;
  for (int t = 0; t < L_SEQ; ++t) msum += m[t];
  float td = 0.f, tp = 0.f;
  for (int t = 0; t < L_SEQ; ++t) {
    float mm = m[t];
    td = fmaf(x[(size_t)t * E_DIM + d], mm, td);
    tp = fmaf(pe[(size_t)t * E_DIM + d], mm, tp);
  }
  td /= msum; tp /= msum;
  for (int t = 0; t < L_SEQ; ++t) {
    float xv = x[(size_t)t * E_DIM + d];
    float wv = xv + pe[(size_t)t * E_DIM + d];
    float tv = fmaf((float)(t + 1), tp, td);
    words[(size_t)t * E_DIM + d]    = wv;
    tgts[(size_t)t * E_DIM + d]     = tv;
    sents_bf[(size_t)t * E_DIM + d] = f2bf_bits(xv);
    words_bf[(size_t)t * E_DIM + d] = f2bf_bits(wv);
  }
  h[d] = 0.f; c[d] = 0.f; hbf[d] = 0;
}

// ---------------- final: classifier + losses ------------------------------------
__global__ void k_final(const float* __restrict__ h,
                        const float* __restrict__ W_s2l, const float* __restrict__ b_s2l,
                        const int* __restrict__ labels, const float* __restrict__ parse_w,
                        const float* __restrict__ ps, const float* __restrict__ acts,
                        float* __restrict__ out, int out_size) {
  __shared__ float red[BS];
  __shared__ float logit[NLAB];
  const int tid = threadIdx.x;
  const int r = tid >> 6, l = tid & 63;
  float s = 0.f;
  if (r < NLAB) {
    const float* Wr = W_s2l + (size_t)r * H_DIM;
    for (int j = l; j < H_DIM; j += 64) s = fmaf(Wr[j], h[j], s);
  }
  red[tid] = s;
  __syncthreads();
  if (tid < NLAB) {
    float t2 = 0.f;
    for (int k = 0; k < 64; ++k) t2 += red[tid * 64 + k];
    logit[tid] = t2 + b_s2l[tid];
  }
  __syncthreads();
  if (tid == 0) {
    float mx = fmaxf(logit[0], fmaxf(logit[1], logit[2]));
    float lse = logf(expf(logit[0] - mx) + expf(logit[1] - mx) + expf(logit[2] - mx)) + mx;
    float pr[NLAB];
    for (int i = 0; i < NLAB; ++i) pr[i] = logit[i] - lse;
    int lab = labels[0]; if (lab < 0) lab = 0; if (lab >= NLAB) lab = NLAB - 1;
    float cls = -pr[lab];
    float nll = 0.f, asum = 0.f, pwsum = 0.f;
    for (int t = 0; t < L_SEQ; ++t) {
      int a = (int)acts[t];
      nll += logf(ps[2 * t + a]);
      asum += acts[t];
      pwsum = fmaf(acts[t], parse_w[t], pwsum);
    }
    float act_nll = -nll / (float)L_SEQ;
    float reward = -cls + ((float)L_SEQ - asum) * 0.5f + pwsum;
    float aloss = act_nll * reward;
    // out = [pred(3), acts(100), action_loss, cls_loss]
    for (int i = 0; i < NLAB && i < out_size; ++i) out[i] = pr[i];
    for (int t = 0; t < L_SEQ && (NLAB + t) < out_size; ++t) out[NLAB + t] = acts[t];
    if (NLAB + L_SEQ     < out_size) out[NLAB + L_SEQ]     = aloss;
    if (NLAB + L_SEQ + 1 < out_size) out[NLAB + L_SEQ + 1] = cls;
  }
}

extern "C" void kernel_launch(void* const* d_in, const int* in_sizes, int n_in,
                              void* d_out, int out_size, void* d_ws, size_t ws_size,
                              hipStream_t stream) {
  const float* sents  = (const float*)d_in[0];
  const float* masks  = (const float*)d_in[1];
  const int*   labels = (const int*)  d_in[2];
  const float* parsew = (const float*)d_in[3];
  const float* post   = (const float*)d_in[4];
  const float* W_ih   = (const float*)d_in[5];
  const float* W_hh   = (const float*)d_in[6];
  const float* b_ih   = (const float*)d_in[7];
  const float* b_hh   = (const float*)d_in[8];
  const float* W_s2t  = (const float*)d_in[9];
  const float* b_s2t  = (const float*)d_in[10];
  const float* W_t2o  = (const float*)d_in[11];
  const float* b_t2o  = (const float*)d_in[12];
  const float* W_s2l  = (const float*)d_in[13];
  const float* b_s2l  = (const float*)d_in[14];

  // workspace carve-up (256B aligned blocks)
  char* base = (char*)d_ws;
  size_t off = 0;
  auto carve = [&](size_t bytes) -> void* {
    void* q = base + off;
    off = (off + bytes + 255) & ~(size_t)255;
    return q;
  };
  unsigned short* wih      = (unsigned short*)carve((size_t)G_DIM * E_DIM * 2);
  unsigned short* whh      = (unsigned short*)carve((size_t)G_DIM * H_DIM * 2);
  unsigned short* sents_bf = (unsigned short*)carve((size_t)L_SEQ * E_DIM * 2);
  unsigned short* words_bf = (unsigned short*)carve((size_t)L_SEQ * E_DIM * 2);
  unsigned short* hbf      = (unsigned short*)carve(H_DIM * 2);
  float* h     = (float*)carve(H_DIM * 4);
  float* c     = (float*)carve(H_DIM * 4);
  float* gates = (float*)carve(G_DIM * 4);
  float* words = (float*)carve((size_t)L_SEQ * E_DIM * 4);
  float* tgts  = (float*)carve((size_t)L_SEQ * E_DIM * 4);
  float* ps    = (float*)carve(L_SEQ * 2 * 4);
  float* acts  = (float*)carve(L_SEQ * 4);
  unsigned* ctrl = (unsigned*)carve(64);
  if (off > ws_size) return;   // scratch too small; nothing safe to do

  hipMemsetAsync(ctrl, 0, 64, stream);
  k_convert<<<2048, BS, 0, stream>>>(W_ih, W_hh, wih, whh);
  k_prepvec<<<(E_DIM + BS - 1) / BS, BS, 0, stream>>>(sents, masks, post, words, tgts,
                                                      sents_bf, words_bf, h, c, hbf);
  k_persist<<<NB, BS, 0, stream>>>(wih, whh, h, c, hbf, gates,
                                   sents_bf, words_bf, words, tgts, b_ih, b_hh,
                                   W_s2t, b_s2t, W_t2o, b_t2o, ps, acts, ctrl);
  k_final<<<1, BS, 0, stream>>>(h, W_s2l, b_s2l, labels, parsew, ps, acts,
                                (float*)d_out, out_size);
}
